// modeler_65068754534665
// MI455X (gfx1250) — matrix operand order repaired and checked by reading, not compiled, not run on hardware
//
#include <hip/hip_runtime.h>
#include <hip/hip_bf16.h>

// ---------------- problem constants (match reference) ----------------
#define NAn  100000
#define NBn  100000
#define FTd  128
#define NE   1600000

typedef __attribute__((ext_vector_type(16))) __bf16 v16bf;
typedef __attribute__((ext_vector_type(8)))  float  v8f;

// Device-scope f32 atomic add (L2-resident RMW; SpMM scatter).
__device__ __forceinline__ void gatom_add_f32(float* p, float v) {
    asm volatile("global_atomic_add_f32 %0, %1, off scope:SCOPE_DEV"
                 :: "v"(p), "v"(v) : "memory");
}

// ---------------- zero-fill ----------------
__global__ void k_zero(float* __restrict__ p, int n4) {
    int i = blockIdx.x * blockDim.x + threadIdx.x;
    if (i < n4) ((float4*)p)[i] = make_float4(0.f, 0.f, 0.f, 0.f);
}

// ---------------- SpMM: Y[row] += val * X[col], D=128 ----------------
// One wave per edge; each lane moves a float4 (512B/edge, fully coalesced).
__global__ void k_spmm(const int* __restrict__ rows, const int* __restrict__ cols,
                       const float* __restrict__ vals, const float* __restrict__ X,
                       float* __restrict__ Y) {
    int t    = blockIdx.x * blockDim.x + threadIdx.x;
    int e    = t >> 5;
    int lane = t & 31;
    if (e >= NE) return;
    int   r = rows[e];
    int   c = cols[e];
    float v = vals[e];
    const float4 x = *(const float4*)(X + (size_t)c * 128 + lane * 4);
    float* y = Y + (size_t)r * 128 + lane * 4;
    gatom_add_f32(y + 0, v * x.x);
    gatom_add_f32(y + 1, v * x.y);
    gatom_add_f32(y + 2, v * x.z);
    gatom_add_f32(y + 3, v * x.w);
}

// ---------------- weight pre-pack: f32 [KIN,KOUT] -> bf16 B-fragments ----------------
// Layout: [nt][kc][lane][16 bf16]; a wave's B operand is then one contiguous
// 32-byte v16bf load per K-chunk (ISA 7.12.2: lane holds column n=lr, K=half*8+j for j<8, 16+half*8+(j-8) for j>=8).
template<int KIN, int KOUT>
__global__ void k_pack_w(const float* __restrict__ W, __bf16* __restrict__ P) {
    int g = blockIdx.x * blockDim.x + threadIdx.x;      // one 16-elem group per thread
    const int KC = KIN / 32;
    const int total = (KOUT / 16) * KC * 32;
    if (g >= total) return;
    int lane = g & 31;
    int kc   = (g >> 5) % KC;
    int nt   = (g >> 5) / KC;
    int half = lane >> 4, lr = lane & 15;
    __bf16* out = P + (size_t)g * 16;
    #pragma unroll
    for (int j = 0; j < 16; ++j)
        out[j] = (__bf16)W[(size_t)(kc * 32 + (j < 8 ? half * 8 + j : 16 + half * 8 + (j - 8))) * KOUT + nt * 16 + lr];
}

// ---------------- bf16 A-fragment loader (ISA 7.12.2 layout) ----------------
__device__ __forceinline__ void ld_a8(v16bf& a, int idx, const float* p) {
    const float4* q = (const float4*)p;
    float4 x0 = q[0], x1 = q[1];
    a[idx + 0] = (__bf16)x0.x; a[idx + 1] = (__bf16)x0.y;
    a[idx + 2] = (__bf16)x0.z; a[idx + 3] = (__bf16)x0.w;
    a[idx + 4] = (__bf16)x1.x; a[idx + 5] = (__bf16)x1.y;
    a[idx + 6] = (__bf16)x1.z; a[idx + 7] = (__bf16)x1.w;
}

// ---------------- dense GEMM: C = act(A[M,KIN] @ W + bias), W pre-packed bf16 ----------------
// 8 waves/block, one 16x16 tile per wave via v_wmma_f32_16x16x32_bf16.
template<int KIN, int KOUT, bool LEAKY>
__global__ void k_gemm_wmma(const float* __restrict__ A, const __bf16* __restrict__ P,
                            const float* __restrict__ bias, float* __restrict__ C, int M) {
    const int wave = threadIdx.x >> 5;
    const int lane = threadIdx.x & 31;
    const int half = lane >> 4;
    const int lr   = lane & 15;
    const int mt   = blockIdx.x * 8 + wave;
    const int nt   = blockIdx.y;
    if (mt * 16 >= M) return;
    const int m0 = mt * 16;
    const int KC = KIN / 32;

    v8f acc = {};
    const float*  arow = A + (size_t)(m0 + lr) * KIN;
    const __bf16* brow = P + ((size_t)nt * KC * 32 + lane) * 16;
    #pragma unroll
    for (int kc = 0; kc < KC; ++kc) {
        v16bf a;
        ld_a8(a, 0, arow + kc * 32 + half * 8);
        ld_a8(a, 8, arow + kc * 32 + 16 + half * 8);
        const v16bf b = *(const v16bf*)(brow + (size_t)kc * 32 * 16);
        acc = __builtin_amdgcn_wmma_f32_16x16x32_bf16(
                false, a, false, b, (short)0, acc, false, false);
    }
    float bv = bias[nt * 16 + lr];
    #pragma unroll
    for (int i = 0; i < 8; ++i) {
        int   row = m0 + half * 8 + i;            // C layout: M = vgpr + 8*half
        float v   = acc[i] + bv;
        if (LEAKY) v = v > 0.f ? v : 0.01f * v;
        C[(size_t)row * KOUT + nt * 16 + lr] = v;
    }
}

// ---------------- fused final FC: C = [U(64) | F(128)] @ W[192,64] + bias ----------------
__global__ void k_fc_wmma(const float* __restrict__ U, const float* __restrict__ F,
                          const __bf16* __restrict__ P, const float* __restrict__ bias,
                          float* __restrict__ C, int M) {
    const int wave = threadIdx.x >> 5;
    const int lane = threadIdx.x & 31;
    const int half = lane >> 4;
    const int lr   = lane & 15;
    const int mt   = blockIdx.x * 8 + wave;
    const int nt   = blockIdx.y;
    if (mt * 16 >= M) return;
    const int m0 = mt * 16;
    const int KC = 192 / 32;

    const float*  bu   = U + (size_t)(m0 + lr) * 64;
    const float*  bf   = F + (size_t)(m0 + lr) * 128;
    const __bf16* brow = P + ((size_t)nt * KC * 32 + lane) * 16;
    v8f acc = {};
    #pragma unroll
    for (int kc = 0; kc < KC; ++kc) {
        v16bf a;
        int g0 = kc * 32 + half * 8;              // 8-wide groups never straddle k=64
        int g1 = kc * 32 + 16 + half * 8;
        ld_a8(a, 0, (g0 < 64) ? (bu + g0) : (bf + g0 - 64));
        ld_a8(a, 8, (g1 < 64) ? (bu + g1) : (bf + g1 - 64));
        const v16bf b = *(const v16bf*)(brow + (size_t)kc * 32 * 16);
        acc = __builtin_amdgcn_wmma_f32_16x16x32_bf16(
                false, a, false, b, (short)0, acc, false, false);
    }
    float bv = bias[nt * 16 + lr];
    #pragma unroll
    for (int i = 0; i < 8; ++i)
        C[(size_t)(m0 + half * 8 + i) * 64 + nt * 16 + lr] = acc[i] + bv;
}

// ---------------- semantic attention over P=2 relations ----------------
// One wave per node: w_p = tanh(x_p @ W1 + b1) @ W2 ; softmax over p ; blend.
template<int D, int H>
__global__ void k_att(const float* __restrict__ v0, const float* __restrict__ v1,
                      const float* __restrict__ W1, const float* __restrict__ b1,
                      const float* __restrict__ W2, float* __restrict__ outp, int N) {
    int node = blockIdx.x * (blockDim.x >> 5) + (threadIdx.x >> 5);
    int lane = threadIdx.x & 31;
    if (node >= N) return;
    const float* x0 = v0 + (size_t)node * D;
    const float* x1 = v1 + (size_t)node * D;

    float w[2];
    #pragma unroll
    for (int p = 0; p < 2; ++p) {
        const float* x = p ? x1 : x0;
        float t = 0.f;
        if (lane < H) {
            float s = b1[lane];
            for (int k = 0; k < D; ++k) s += x[k] * W1[k * H + lane];
            t = tanhf(s) * W2[lane];
        }
        #pragma unroll
        for (int off = 16; off; off >>= 1) t += __shfl_xor(t, off);
        w[p] = t;
    }
    float m  = fmaxf(w[0], w[1]);
    float e0 = __expf(w[0] - m), e1 = __expf(w[1] - m);
    float inv = 1.f / (e0 + e1);
    float beta0 = e0 * inv, beta1 = e1 * inv;
    for (int c = lane; c < D; c += 32)
        outp[(size_t)node * D + c] = beta0 * x0[c] + beta1 * x1[c];
}

// ---------------- orchestration ----------------
extern "C" void kernel_launch(void* const* d_in, const int* in_sizes, int n_in,
                              void* d_out, int out_size, void* d_ws, size_t ws_size,
                              hipStream_t stream) {
    (void)in_sizes; (void)n_in; (void)out_size; (void)ws_size;
    const float* features = (const float*)d_in[0];
    const int *row_aa=(const int*)d_in[1], *col_aa=(const int*)d_in[2];
    const float* val_aa=(const float*)d_in[3];
    const int *row_ab=(const int*)d_in[4], *col_ab=(const int*)d_in[5];
    const float* val_ab=(const float*)d_in[6];
    const int *row_ba=(const int*)d_in[7], *col_ba=(const int*)d_in[8];
    const float* val_ba=(const float*)d_in[9];
    const int *row_bb=(const int*)d_in[10], *col_bb=(const int*)d_in[11];
    const float* val_bb=(const float*)d_in[12];
    const float *W0_aa=(const float*)d_in[13], *b0_aa=(const float*)d_in[14];
    const float *W1_aa=(const float*)d_in[15], *b1_aa=(const float*)d_in[16];
    const float *W0_ab=(const float*)d_in[17], *b0_ab=(const float*)d_in[18];
    const float *W1_ab=(const float*)d_in[19], *b1_ab=(const float*)d_in[20];
    const float *W0_ba=(const float*)d_in[21], *b0_ba=(const float*)d_in[22];
    const float *W1_ba=(const float*)d_in[23], *b1_ba=(const float*)d_in[24];
    const float *W0_bb=(const float*)d_in[25], *b0_bb=(const float*)d_in[26];
    const float *W1_bb=(const float*)d_in[27], *b1_bb=(const float*)d_in[28];
    const float *att0a_W1=(const float*)d_in[29], *att0a_b1=(const float*)d_in[30], *att0a_W2=(const float*)d_in[31];
    const float *att1a_W1=(const float*)d_in[32], *att1a_b1=(const float*)d_in[33], *att1a_W2=(const float*)d_in[34];
    const float *Wf_a=(const float*)d_in[35], *bf_a=(const float*)d_in[36];
    const float *att0b_W1=(const float*)d_in[37], *att0b_b1=(const float*)d_in[38], *att0b_W2=(const float*)d_in[39];
    const float *att1b_W1=(const float*)d_in[40], *att1b_b1=(const float*)d_in[41], *att1b_W2=(const float*)d_in[42];
    const float *Wf_b=(const float*)d_in[43], *bf_b=(const float*)d_in[44];

    const float* fa = features;
    const float* fb = features + (size_t)NAn * FTd;
    float* out = (float*)d_out;

    // workspace layout
    const size_t S = (size_t)NAn * 128;
    float* ws  = (float*)d_ws;
    float* B0  = ws;            // spmm temp rel0
    float* B1  = ws + 1 * S;    // spmm temp rel1
    float* B2  = ws + 2 * S;    // gcn out rel0
    float* B3  = ws + 3 * S;    // gcn out rel1
    float* E1A = ws + 4 * S;    // layer-0 embedding, type a
    float* E1B = ws + 5 * S;    // layer-0 embedding, type b
    float* Ubuf= ws + 6 * S;    // 64-wide layer-1 summary
    // packed bf16 weights (fragment layout), after Ubuf (NAn*64 floats)
    __bf16* pk = (__bf16*)(Ubuf + (size_t)NAn * 64);
    const size_t SZ0 = 128 * 128, SZ1 = 128 * 64, SZF = 192 * 64;
    __bf16 *P0aa = pk,              *P0ab = P0aa + SZ0, *P0ba = P0ab + SZ0, *P0bb = P0ba + SZ0;
    __bf16 *P1aa = P0bb + SZ0,      *P1ab = P1aa + SZ1, *P1ba = P1ab + SZ1, *P1bb = P1ba + SZ1;
    __bf16 *PFa  = P1bb + SZ1,      *PFb  = PFa + SZF;

    auto zero = [&](float* p, size_t n) {
        int n4 = (int)(n / 4);
        k_zero<<<(n4 + 255) / 256, 256, 0, stream>>>(p, n4);
    };
    auto spmm = [&](const int* r, const int* c, const float* v,
                    const float* X, float* Y) {
        zero(Y, S);
        k_spmm<<<(NE * 32) / 256, 256, 0, stream>>>(r, c, v, X, Y);
    };

    // ---- one-time weight packing (tiny) ----
    k_pack_w<128,128><<<(16384/16 + 255)/256, 256, 0, stream>>>(W0_aa, P0aa);
    k_pack_w<128,128><<<(16384/16 + 255)/256, 256, 0, stream>>>(W0_ab, P0ab);
    k_pack_w<128,128><<<(16384/16 + 255)/256, 256, 0, stream>>>(W0_ba, P0ba);
    k_pack_w<128,128><<<(16384/16 + 255)/256, 256, 0, stream>>>(W0_bb, P0bb);
    k_pack_w<128,64 ><<<(8192/16  + 255)/256, 256, 0, stream>>>(W1_aa, P1aa);
    k_pack_w<128,64 ><<<(8192/16  + 255)/256, 256, 0, stream>>>(W1_ab, P1ab);
    k_pack_w<128,64 ><<<(8192/16  + 255)/256, 256, 0, stream>>>(W1_ba, P1ba);
    k_pack_w<128,64 ><<<(8192/16  + 255)/256, 256, 0, stream>>>(W1_bb, P1bb);
    k_pack_w<192,64 ><<<(12288/16 + 255)/256, 256, 0, stream>>>(Wf_a,  PFa);
    k_pack_w<192,64 ><<<(12288/16 + 255)/256, 256, 0, stream>>>(Wf_b,  PFb);

    const dim3 blk(256);                           // 8 waves/block
    const dim3 g128((NAn / 16 + 7) / 8, 128 / 16); // KOUT=128
    const dim3 g64 ((NAn / 16 + 7) / 8, 64 / 16);  // KOUT=64
    const int  attBlocks = (NAn + 7) / 8;
    const dim3 ablk(256);

    // ---- layer 0, type a ----
    spmm(row_aa, col_aa, val_aa, fa, B0);
    spmm(row_ab, col_ab, val_ab, fb, B1);
    k_gemm_wmma<128,128,true><<<g128, blk, 0, stream>>>(B0, P0aa, b0_aa, B2, NAn);
    k_gemm_wmma<128,128,true><<<g128, blk, 0, stream>>>(B1, P0ab, b0_ab, B3, NAn);
    k_att<128,32><<<attBlocks, ablk, 0, stream>>>(B2, B3, att0a_W1, att0a_b1, att0a_W2, E1A, NAn);

    // ---- layer 0, type b ----
    spmm(row_ba, col_ba, val_ba, fa, B0);
    spmm(row_bb, col_bb, val_bb, fb, B1);
    k_gemm_wmma<128,128,true><<<g128, blk, 0, stream>>>(B0, P0ba, b0_ba, B2, NBn);
    k_gemm_wmma<128,128,true><<<g128, blk, 0, stream>>>(B1, P0bb, b0_bb, B3, NBn);
    k_att<128,32><<<attBlocks, ablk, 0, stream>>>(B2, B3, att0b_W1, att0b_b1, att0b_W2, E1B, NBn);

    // ---- layer 1, type a ----
    spmm(row_aa, col_aa, val_aa, E1A, B0);
    spmm(row_ab, col_ab, val_ab, E1B, B1);
    k_gemm_wmma<128,64,true><<<g64, blk, 0, stream>>>(B0, P1aa, b1_aa, B2, NAn);
    k_gemm_wmma<128,64,true><<<g64, blk, 0, stream>>>(B1, P1ab, b1_ab, B3, NAn);
    k_att<64,16><<<attBlocks, ablk, 0, stream>>>(B2, B3, att1a_W1, att1a_b1, att1a_W2, Ubuf, NAn);
    k_fc_wmma<<<g64, blk, 0, stream>>>(Ubuf, fa, PFa, bf_a, out, NAn);

    // ---- layer 1, type b ----
    spmm(row_ba, col_ba, val_ba, E1A, B0);
    spmm(row_bb, col_bb, val_bb, E1B, B1);
    k_gemm_wmma<128,64,true><<<g64, blk, 0, stream>>>(B0, P1ba, b1_ba, B2, NBn);
    k_gemm_wmma<128,64,true><<<g64, blk, 0, stream>>>(B1, P1bb, b1_bb, B3, NBn);
    k_att<64,16><<<attBlocks, ablk, 0, stream>>>(B2, B3, att1b_W1, att1b_b1, att1b_W2, Ubuf, NBn);
    k_fc_wmma<<<g64, blk, 0, stream>>>(Ubuf, fb, PFb, bf_b, out + (size_t)NAn * 64, NBn);
}
